// AutoencoderExcitedMACE_36825049596262
// MI455X (gfx1250) — matrix operand order repaired and checked by reading, not compiled, not run on hardware
//
#include <hip/hip_runtime.h>
#include <hip/hip_bf16.h>
#include <math.h>

// ---------------------------------------------------------------------------
// MACE-style GNN forward for MI455X (gfx1250, wave32, WMMA).
// Heavy GEMM work (radial MLP E x (8->64->64->288), node up-projection) runs
// on v_wmma_f32_16x16x32_f16. Edge-parallel fused kernel avoids materializing
// the 295MB Rw tensor; scatter into agg via f32 atomics.
// Weights staged TRANSPOSED in LDS so B-fragments load as contiguous
// ds_load_b128 pairs; A-fragments relayout via 16B-aligned v8h LDS loads.
// ---------------------------------------------------------------------------

typedef _Float16 half_t;
typedef __attribute__((ext_vector_type(8)))  _Float16 v8h;
typedef __attribute__((ext_vector_type(16))) _Float16 v16h;
typedef __attribute__((ext_vector_type(8)))  float    v8f;

#define NNODE 16000
#define NEDGE 256000
#define NGRAPH 64
#define NELEM 10
#define CCH 32
#define NSH 9
#define FROW 288   // NSH*CCH

__device__ __forceinline__ float silu_f(float x) {
    return x / (1.0f + __expf(-x));
}

__device__ __forceinline__ v8f wmma_f16(v16h a, v16h b, v8f c) {
    // D = A(16x32 f16) * B(32x16 f16) + C(16x16 f32)
    return __builtin_amdgcn_wmma_f32_16x16x32_f16(
        false, a, false, b, (short)0, c, false, false);
}

__device__ __forceinline__ v16h cat8(v8h lo, v8h hi) {
    return __builtin_shufflevector(lo, hi, 0, 1, 2, 3, 4, 5, 6, 7,
                                           8, 9, 10, 11, 12, 13, 14, 15);
}

// A-fragment (ISA 7.12.2, 16-bit A 16x32): lane 0-15 K={0..7,16..23},
// lane 16-31 K={8..15,24..31}  ->  two contiguous 8-half runs at
// offsets 8*hl and 16+8*hl within the 32-wide K row.
__device__ __forceinline__ v16h a_frag_lds(const half_t* row, int hl) {
    v8h lo = *(const v8h*)(row + 8 * hl);
    v8h hi = *(const v8h*)(row + 16 + 8 * hl);
    return cat8(lo, hi);
}
// B-fragment: same K order as A (lane 0-15 K={0..7,16..23}, lane 16-31
// K={8..15,24..31}) -> two 8-half runs in a transposed [n][k] tile.
__device__ __forceinline__ v16h b_frag_lds(const half_t* rowT, int hl) {
    v8h lo = *(const v8h*)(rowT + 8 * hl);
    v8h hi = *(const v8h*)(rowT + 16 + 8 * hl);
    return cat8(lo, hi);
}

// ---------------------------------------------------------------------------
__global__ void k_zero(float* p, int n) {
    int i = blockIdx.x * 256 + threadIdx.x;
    if (i < n) p[i] = 0.0f;
}

// ---------------------------------------------------------------------------
// Per-edge geometry: distance, spherical harmonics (9), Bessel basis * envelope
// stored as f16 (WMMA A operand for the radial MLP).
__global__ void k_prep_edges(const float* __restrict__ pos,
                             const float* __restrict__ shifts,
                             const int* __restrict__ eidx,
                             float* __restrict__ shE,
                             half_t* __restrict__ ef16, int Ecnt) {
    int e = blockIdx.x * 256 + threadIdx.x;
    if (e >= Ecnt) return;
    int s = eidx[e], d = eidx[Ecnt + e];
    float vx = pos[s * 3 + 0] - pos[d * 3 + 0] + shifts[e * 3 + 0];
    float vy = pos[s * 3 + 1] - pos[d * 3 + 1] + shifts[e * 3 + 1];
    float vz = pos[s * 3 + 2] - pos[d * 3 + 2] + shifts[e * 3 + 2];
    float r = sqrtf(vx * vx + vy * vy + vz * vz + 1e-12f);
    float ir = 1.0f / r;
    float x = vx * ir, y = vy * ir, z = vz * ir;
    const float s3 = 1.7320508075688772f;
    const float s5 = 2.2360679774997896f;
    const float s15 = 3.872983346207417f;
    float* o = shE + (size_t)e * NSH;
    o[0] = 1.0f;
    o[1] = s3 * x;  o[2] = s3 * y;  o[3] = s3 * z;
    o[4] = s15 * x * y; o[5] = s15 * y * z;
    o[6] = 0.5f * s5 * (3.0f * z * z - 1.0f);
    o[7] = s15 * x * z;
    o[8] = 0.5f * s15 * (x * x - y * y);
    float u = r * 0.2f;                          // r / R_MAX
    float u2 = u * u, u3 = u2 * u;
    float u6 = u3 * u3, u7 = u6 * u, u8 = u7 * u;
    float env = 1.0f - 28.0f * u6 + 48.0f * u7 - 21.0f * u8;  // P=6 polynomial
    env = (u < 1.0f) ? env : 0.0f;
    float base = 0.6324555320336759f * ir * env;  // sqrt(2/R_MAX)/r * env
    const float PI = 3.14159265358979323846f;
#pragma unroll
    for (int i = 0; i < 8; i++)
        ef16[(size_t)e * 8 + i] = (half_t)(base * __sinf((float)(i + 1) * PI * u));
}

// ---------------------------------------------------------------------------
// Node init: feats[:,0,:] = node_attrs @ W_embed, rest zero; element index
// (argmax of one-hot) + value; e0 segment sum.
__global__ void k_node_init(const float* __restrict__ attrs,
                            const float* __restrict__ Wemb,
                            const float* __restrict__ E0v,
                            const int* __restrict__ batch,
                            float* __restrict__ feats,
                            int* __restrict__ elem, float* __restrict__ elemval,
                            float* __restrict__ e0g, int Ncnt) {
    int n = blockIdx.x * 256 + threadIdx.x;
    if (n >= Ncnt) return;
    const float* a = attrs + (size_t)n * NELEM;
    float best = -1e30f; int bi = 0; float e0 = 0.0f;
#pragma unroll
    for (int k = 0; k < NELEM; k++) {
        float v = a[k];
        e0 += v * E0v[k];
        if (v > best) { best = v; bi = k; }
    }
    elem[n] = bi; elemval[n] = best;
    atomicAdd(&e0g[batch[n]], e0);
    for (int c = 0; c < CCH; c++) {
        float acc = 0.0f;
#pragma unroll
        for (int k = 0; k < NELEM; k++) acc += a[k] * Wemb[k * CCH + c];
        feats[(size_t)n * FROW + c] = acc;
    }
    for (int i = CCH; i < FROW; i++) feats[(size_t)n * FROW + i] = 0.0f;
}

// ---------------------------------------------------------------------------
// up = feats[:,0,:] @ W_up[l]  (N x 32 @ 32 x 32) via WMMA; one wave = 16 rows.
__global__ void __launch_bounds__(128) k_up(const float* __restrict__ feats,
                                            const float* __restrict__ Wup,
                                            float* __restrict__ up, int Ncnt) {
    const int lane = threadIdx.x & 31;
    const int wave = threadIdx.x >> 5;
    const int nl = lane & 15, hl = lane >> 4;
    const int t = blockIdx.x * 4 + wave;
    const int ntiles = Ncnt / 16;
    if (t >= ntiles) return;  // wave-uniform
    v16h a;
    const int row = t * 16 + nl;
#pragma unroll
    for (int i = 0; i < 16; i++) {
        int k = (i & 7) + 8 * hl + ((i >> 3) << 4);
        a[i] = (half_t)feats[(size_t)row * FROW + k];
    }
#pragma unroll
    for (int nt = 0; nt < 2; nt++) {
        v16h b;
#pragma unroll
        for (int i = 0; i < 16; i++) {
            int k = (i & 7) + 8 * hl + ((i >> 3) << 4);
            b[i] = (half_t)Wup[k * CCH + nt * 16 + nl];
        }
        v8f acc = {};
        acc = wmma_f16(a, b, acc);
#pragma unroll
        for (int j = 0; j < 8; j++)
            up[(size_t)(t * 16 + j + 8 * hl) * CCH + nt * 16 + nl] = acc[j];
    }
}

// ---------------------------------------------------------------------------
// Fused radial MLP + message + scatter. One wave handles 16 edges:
//   h1 = silu(ef @ Wr1)        K=8 (zero-padded to 32), 4 n-tiles
//   h2 = silu(h1 @ Wr2)        K=64 -> 2 WMMA per n-tile, 4 n-tiles
//   Rw = h2 @ Wr3              K=64 -> 2 WMMA per n-tile, 18 n-tiles
//   agg[dst] += Rw * sh * up[src] / 16     (f32 atomics)
// Weights live transposed ([n][k]) in LDS -> B-fragments are 2x ds_load_b128.
// C-fragment -> A-fragment relayout through a wave-private LDS tile.
__global__ void __launch_bounds__(128) k_radial_msg(
    const half_t* __restrict__ ef16, const float* __restrict__ shE,
    const float* __restrict__ up, const int* __restrict__ eidx,
    const float* __restrict__ Wr1g, const float* __restrict__ Wr2g,
    const float* __restrict__ Wr3g, float* __restrict__ agg, int Ecnt) {
    __shared__ half_t sWr1T[64][8];        //  1 KB  [n][k], k<8
    __shared__ half_t sWr2T[64][64];       //  8 KB  [n][k]
    __shared__ half_t sWr3T[288][64];      // 36 KB  [n][k]
    __shared__ half_t hbuf[4][2][16][72];  // 18 KB wave-private staging (padded)

    const int tid = threadIdx.x;
    for (int i = tid; i < 64 * 8; i += 128) {
        int n = i >> 3, k = i & 7;
        sWr1T[n][k] = (half_t)Wr1g[k * 64 + n];
    }
    for (int i = tid; i < 64 * 64; i += 128) {
        int n = i >> 6, k = i & 63;
        sWr2T[n][k] = (half_t)Wr2g[k * 64 + n];
    }
    for (int i = tid; i < 288 * 64; i += 128) {
        int n = i >> 6, k = i & 63;
        sWr3T[n][k] = (half_t)Wr3g[k * 288 + n];
    }
    __syncthreads();

    const int lane = tid & 31;
    const int wave = tid >> 5;
    const int nl = lane & 15, hl = lane >> 4;
    const int slots = gridDim.x * 4;
    const int ntiles = Ecnt / 16;
    half_t (*h1)[72] = hbuf[wave][0];
    half_t (*h2)[72] = hbuf[wave][1];
    const v8h zero8 = {};

    for (int t = blockIdx.x * 4 + wave; t < ntiles; t += slots) {
        if (t + slots < ntiles)  // gfx1250 global_prefetch_b8 on next tile
            __builtin_prefetch(ef16 + (size_t)(t + slots) * 16 * 8, 0, 1);

        // ---- stage 1: A = edge_feats tile (K padded 8 -> 32). Only hl==0
        // lanes carry data (their K slots are 0..7); everything else zero.
        v8h efv = zero8;
        if (hl == 0) efv = *(const v8h*)(ef16 + (size_t)(t * 16 + nl) * 8);
        v16h a = cat8(efv, zero8);
#pragma unroll
        for (int nt = 0; nt < 4; nt++) {
            v8h wv = zero8;
            if (hl == 0) wv = *(const v8h*)&sWr1T[nt * 16 + nl][0];
            v16h b = cat8(wv, zero8);
            v8f acc = {};
            acc = wmma_f16(a, b, acc);
#pragma unroll
            for (int j = 0; j < 8; j++)
                h1[j + 8 * hl][nt * 16 + nl] = (half_t)silu_f(acc[j]);
        }

        // ---- stage 2: h2 = silu(h1 @ Wr2), K=64 --------------------------
        v16h a0 = a_frag_lds(&h1[nl][0], hl);
        v16h a1 = a_frag_lds(&h1[nl][32], hl);
#pragma unroll
        for (int nt = 0; nt < 4; nt++) {
            v16h b0 = b_frag_lds(&sWr2T[nt * 16 + nl][0], hl);
            v16h b1 = b_frag_lds(&sWr2T[nt * 16 + nl][32], hl);
            v8f acc = {};
            acc = wmma_f16(a0, b0, acc);
            acc = wmma_f16(a1, b1, acc);
#pragma unroll
            for (int j = 0; j < 8; j++)
                h2[j + 8 * hl][nt * 16 + nl] = (half_t)silu_f(acc[j]);
        }

        // ---- stage 3: Rw tile-by-tile, fuse msg + scatter ----------------
        a0 = a_frag_lds(&h2[nl][0], hl);
        a1 = a_frag_lds(&h2[nl][32], hl);
        int srcR[8], dstR[8], eR[8];
#pragma unroll
        for (int j = 0; j < 8; j++) {
            int e = t * 16 + j + 8 * hl;   // matches C-fragment row m = j+8*hl
            eR[j] = e;
            srcR[j] = eidx[e];
            dstR[j] = eidx[Ecnt + e];
        }
        float upv[2][8];                   // up[src][c], c = half*16 + nl
#pragma unroll
        for (int j = 0; j < 8; j++) {
            upv[0][j] = up[(size_t)srcR[j] * CCH + nl];
            upv[1][j] = up[(size_t)srcR[j] * CCH + 16 + nl];
        }
        for (int shidx = 0; shidx < NSH; shidx++) {
            float shv[8];
#pragma unroll
            for (int j = 0; j < 8; j++)
                shv[j] = shE[(size_t)eR[j] * NSH + shidx] * 0.0625f;  // /AVG_NBR
#pragma unroll
            for (int half = 0; half < 2; half++) {
                const int col = (shidx * 2 + half) * 16 + nl;  // 0..287
                v16h b0 = b_frag_lds(&sWr3T[col][0], hl);
                v16h b1 = b_frag_lds(&sWr3T[col][32], hl);
                v8f acc = {};
                acc = wmma_f16(a0, b0, acc);
                acc = wmma_f16(a1, b1, acc);
                const int c = half * 16 + nl;
#pragma unroll
                for (int j = 0; j < 8; j++) {
                    float m = acc[j] * shv[j] * upv[half][j];
                    atomicAdd(&agg[(size_t)dstR[j] * FROW + shidx * CCH + c], m);
                }
            }
        }
    }
}

// ---------------------------------------------------------------------------
// new = agg @ W_prod[LVALS] + (one-hot) W_sc contraction; polynomial channel
// correction on the scalar (l=0) block. 8 nodes / block, thread = (node, d).
__global__ void k_mix(const float* __restrict__ agg,
                      const float* __restrict__ Wsc,   // [K][3][32][32] (layer slice)
                      const float* __restrict__ Wprod, // [3][32][32]
                      const float* __restrict__ Wc,    // [3][K][32][32]
                      const int* __restrict__ elem, const float* __restrict__ elemval,
                      float* __restrict__ feats, int Ncnt) {
    __shared__ float fsh[8][NSH][CCH];
    __shared__ float s0[8][CCH];
    const int loc = threadIdx.x >> 5;
    const int d = threadIdx.x & 31;
    const int n = blockIdx.x * 8 + loc;
    const bool ok = n < Ncnt;
    if (ok)
        for (int lh = 0; lh < NSH; lh++)
            fsh[loc][lh][d] = feats[(size_t)n * FROW + lh * CCH + d];
    __syncthreads();
    int el = 0; float ev = 0.0f;
    if (ok) { el = elem[n]; ev = elemval[n]; }
    if (ok) {
        const float* wsc_e = Wsc + (size_t)el * 3 * 1024;
        float news0 = 0.0f;
        for (int lh = 0; lh < NSH; lh++) {
            int lv = (lh == 0) ? 0 : ((lh < 4) ? 1 : 2);  // LVALS
            const float* wp = Wprod + lv * 1024;
            const float* ws = wsc_e + lv * 1024;
            const float* ag = agg + (size_t)n * FROW + lh * CCH;
            float acc = 0.0f;
            for (int c = 0; c < CCH; c++)
                acc += ag[c] * wp[c * CCH + d] + ev * fsh[loc][lh][c] * ws[c * CCH + d];
            if (lh == 0) news0 = acc;
            else feats[(size_t)n * FROW + lh * CCH + d] = acc;
        }
        s0[loc][d] = news0;
    }
    __syncthreads();
    if (ok) {
        const float* w0 = Wc + (size_t)(0 * NELEM + el) * 1024;
        const float* w1 = Wc + (size_t)(1 * NELEM + el) * 1024;
        const float* w2 = Wc + (size_t)(2 * NELEM + el) * 1024;
        float acc = 0.0f;
        for (int c = 0; c < CCH; c++) {
            float x = s0[loc][c];
            acc += x * (w0[c * CCH + d] + x * (w1[c * CCH + d] + x * w2[c * CCH + d]));
        }
        feats[(size_t)n * FROW + d] = s0[loc][d] + ev * acc;
    }
}

// ---------------------------------------------------------------------------
// Per-node readouts -> per-graph sums (f32 atomics).
__global__ void k_readout(const float* __restrict__ feats, const int* __restrict__ batch,
                          const float* __restrict__ We1, const float* __restrict__ We2,
                          const float* __restrict__ Wi1, const float* __restrict__ Wi2,
                          const float* __restrict__ Wdp,
                          float* __restrict__ eng, float* __restrict__ inv,
                          float* __restrict__ dip, int Ncnt) {
    int n = blockIdx.x * 256 + threadIdx.x;
    if (n >= Ncnt) return;
    int g = batch[n];
    const float* s = feats + (size_t)n * FROW;
    float h[16];
#pragma unroll
    for (int j = 0; j < 16; j++) {
        float acc = 0.0f;
        for (int c = 0; c < CCH; c++) acc += s[c] * We1[c * 16 + j];
        h[j] = silu_f(acc);
    }
#pragma unroll
    for (int o = 0; o < 3; o++) {
        float acc = 0.0f;
        for (int j = 0; j < 16; j++) acc += h[j] * We2[j * 3 + o];
        atomicAdd(&eng[g * 3 + o], acc);
    }
#pragma unroll
    for (int j = 0; j < 16; j++) {
        float acc = 0.0f;
        for (int c = 0; c < CCH; c++) acc += s[c] * Wi1[c * 16 + j];
        h[j] = silu_f(acc);
    }
#pragma unroll
    for (int o = 0; o < 4; o++) {
        float acc = 0.0f;
        for (int j = 0; j < 16; j++) acc += h[j] * Wi2[j * 4 + o];
        atomicAdd(&inv[g * 4 + o], acc);
    }
#pragma unroll
    for (int v = 0; v < 3; v++) {
        const float* vv = s + (1 + v) * CCH;
        for (int o = 0; o < 6; o++) {
            float acc = 0.0f;
            for (int c = 0; c < CCH; c++) acc += vv[c] * Wdp[c * 6 + o];
            atomicAdd(&dip[g * 18 + v * 6 + o], acc);
        }
    }
}

// ---------------------------------------------------------------------------
// Final per-graph decode + output assembly (G=64 rows x 24 cols).
__global__ void k_decode(const float* __restrict__ eng, const float* __restrict__ inv,
                         const float* __restrict__ dip, const float* __restrict__ e0g,
                         const float* __restrict__ Wd1, const float* __restrict__ bd1,
                         const float* __restrict__ Wd2, const float* __restrict__ bd2,
                         float* __restrict__ out) {
    int g = threadIdx.x;
    if (g >= NGRAPH) return;
    float h[64];
    for (int j = 0; j < 64; j++) {
        float acc = bd1[j];
#pragma unroll
        for (int p = 0; p < 4; p++) acc += inv[g * 4 + p] * Wd1[p * 64 + j];
        h[j] = silu_f(acc);
    }
    float e0 = e0g[g];
#pragma unroll
    for (int o = 0; o < 3; o++) {
        float acc = bd2[o];
        for (int j = 0; j < 64; j++) acc += h[j] * Wd2[j * 3 + o];
        out[g * 24 + o] = acc + e0;                 // decoded
        out[g * 24 + 3 + o] = eng[g * 3 + o] + e0;  // total_e
    }
#pragma unroll
    for (int dd = 0; dd < 6; dd++)
#pragma unroll
        for (int v = 0; v < 3; v++)
            out[g * 24 + 6 + dd * 3 + v] = dip[g * 18 + v * 6 + dd];  // (0,2,1) transpose
}

// ---------------------------------------------------------------------------
extern "C" void kernel_launch(void* const* d_in, const int* in_sizes, int n_in,
                              void* d_out, int out_size, void* d_ws, size_t ws_size,
                              hipStream_t stream) {
    (void)in_sizes; (void)n_in; (void)out_size; (void)ws_size;
    const int N = NNODE, E = NEDGE, G = NGRAPH;

    const float* positions  = (const float*)d_in[0];
    const float* node_attrs = (const float*)d_in[1];
    const float* shifts     = (const float*)d_in[2];
    const float* E0v        = (const float*)d_in[3];
    const float* W_embed    = (const float*)d_in[4];
    const float* W_up       = (const float*)d_in[5];
    const float* W_r1       = (const float*)d_in[6];
    const float* W_r2       = (const float*)d_in[7];
    const float* W_r3       = (const float*)d_in[8];
    const float* W_sc       = (const float*)d_in[9];
    const float* W_prod     = (const float*)d_in[10];
    const float* Wc         = (const float*)d_in[11];
    const float* We1        = (const float*)d_in[12];
    const float* We2        = (const float*)d_in[13];
    const float* Wi1        = (const float*)d_in[14];
    const float* Wi2        = (const float*)d_in[15];
    const float* Wdp        = (const float*)d_in[16];
    const float* Wd1        = (const float*)d_in[17];
    const float* bd1        = (const float*)d_in[18];
    const float* Wd2        = (const float*)d_in[19];
    const float* bd2        = (const float*)d_in[20];
    const int*   edge_index = (const int*)d_in[21];
    const int*   batch      = (const int*)d_in[22];
    float* out = (float*)d_out;

    // ---- workspace carve-out (~53 MB) -----------------------------------
    char* p = (char*)d_ws;
    auto carve = [&](size_t bytes) -> char* {
        char* r = p;
        p += (bytes + 255) & ~size_t(255);
        return r;
    };
    float*  shE    = (float*)carve((size_t)E * NSH * 4);
    half_t* ef16   = (half_t*)carve((size_t)E * 8 * 2);
    float*  feats  = (float*)carve((size_t)N * FROW * 4);
    float*  aggb   = (float*)carve((size_t)N * FROW * 4);
    float*  upb    = (float*)carve((size_t)N * CCH * 4);
    int*    elem   = (int*)carve((size_t)N * 4);
    float*  elemval= (float*)carve((size_t)N * 4);
    float*  e0g    = (float*)carve(G * 4);
    float*  engb   = (float*)carve(G * 3 * 4);
    float*  invb   = (float*)carve(G * 4 * 4);
    float*  dipb   = (float*)carve(G * 18 * 4);

    // zero accumulators (fresh every call -> graph-replay deterministic)
    k_zero<<<1, 64, 0, stream>>>(e0g, G);
    k_zero<<<1, 192, 0, stream>>>(engb, G * 3);
    k_zero<<<1, 256, 0, stream>>>(invb, G * 4);
    k_zero<<<5, 256, 0, stream>>>(dipb, G * 18);

    k_prep_edges<<<(E + 255) / 256, 256, 0, stream>>>(positions, shifts, edge_index,
                                                      shE, ef16, E);
    k_node_init<<<(N + 255) / 256, 256, 0, stream>>>(node_attrs, W_embed, E0v, batch,
                                                     feats, elem, elemval, e0g, N);

    for (int l = 0; l < 2; l++) {
        k_up<<<(N / 16 + 3) / 4, 128, 0, stream>>>(feats, W_up + (size_t)l * 1024, upb, N);
        k_zero<<<(N * FROW + 255) / 256, 256, 0, stream>>>(aggb, N * FROW);
        k_radial_msg<<<250, 128, 0, stream>>>(ef16, shE, upb, edge_index,
                                              W_r1 + (size_t)l * 8 * 64,
                                              W_r2 + (size_t)l * 64 * 64,
                                              W_r3 + (size_t)l * 64 * 288,
                                              aggb, E);
        k_mix<<<(N + 7) / 8, 256, 0, stream>>>(aggb,
                                               W_sc + (size_t)l * NELEM * 3 * 1024,
                                               W_prod + (size_t)l * 3 * 1024,
                                               Wc + (size_t)l * 3 * NELEM * 1024,
                                               elem, elemval, feats, N);
        k_readout<<<(N + 255) / 256, 256, 0, stream>>>(feats, batch,
                                                       We1 + (size_t)l * 512,
                                                       We2 + (size_t)l * 48,
                                                       Wi1 + (size_t)l * 512,
                                                       Wi2 + (size_t)l * 64,
                                                       Wdp + (size_t)l * 192,
                                                       engb, invb, dipb, N);
    }
    k_decode<<<1, 64, 0, stream>>>(engb, invb, dipb, e0g, Wd1, bd1, Wd2, bd2, out);
}
